// CrossMultiHeadAttention_18726057410880
// MI455X (gfx1250) — matrix operand order repaired and checked by reading, not compiled, not run on hardware
//
#include <hip/hip_runtime.h>
#include <math.h>
#include <stdint.h>

typedef __attribute__((ext_vector_type(16))) __bf16 v16bf;
typedef __attribute__((ext_vector_type(8)))  float  v8f;

#define WMMA_BF16(a, b, c) \
  __builtin_amdgcn_wmma_f32_16x16x32_bf16(false, (a), false, (b), (short)0, (c), false, false)

static constexpr int B_  = 8;
static constexpr int T_  = 2048;
static constexpr int C_  = 192;
static constexpr int H_  = 6;
static constexpr int HS_ = 32;
static constexpr int SK_ = 64;   // keys per attention iteration (2 LDS chunks deep)

#if defined(__has_builtin)
#if __has_builtin(__builtin_amdgcn_global_load_async_to_lds_b64)
#define HAVE_ASYNC_LDS 1
#endif
#endif
#ifndef HAVE_ASYNC_LDS
#define HAVE_ASYNC_LDS 0
#endif

#if HAVE_ASYNC_LDS
// exact param type per hipcc diagnostic: vector_size(8) int, AS(1) / AS(3)
typedef int v2i __attribute__((__vector_size__(8)));
typedef __attribute__((address_space(1))) v2i* gv2ip;
typedef __attribute__((address_space(3))) v2i* lv2ip;
#endif

// ---------- fragment loaders (contiguous k = 16*(lane/16) + e for both A and B, wave32) ----------

// A fragment 16x32 (MxK), src points at (row0, k0) of a row-major bf16 matrix.
__device__ __forceinline__ v16bf load_a_bf16(const __bf16* src, int ld) {
  const int lane = threadIdx.x & 31;
  const int r  = lane & 15;
  const int kb = (lane >> 4) * 16;
  const __bf16* p = src + (size_t)r * ld + kb;
  v16bf a;
#pragma unroll
  for (int e = 0; e < 16; ++e) a[e] = p[e];
  return a;
}

__device__ __forceinline__ v16bf load_a_f32(const float* src, int ld) {
  const int lane = threadIdx.x & 31;
  const int r  = lane & 15;
  const int kb = (lane >> 4) * 16;
  const float* p = src + (size_t)r * ld + kb;
  v16bf a;
#pragma unroll
  for (int e = 0; e < 16; ++e) a[e] = (__bf16)p[e];
  return a;
}

// B fragment 32x16 (KxN), src points at (k0, col0) of a row-major [K x N] matrix.
__device__ __forceinline__ v16bf load_b_bf16(const __bf16* src, int ld) {
  const int lane = threadIdx.x & 31;
  const int n  = lane & 15;
  const int kb = (lane >> 4) * 16;
  const __bf16* p = src + (size_t)kb * ld + n;
  v16bf b;
#pragma unroll
  for (int e = 0; e < 16; ++e) b[e] = p[(size_t)e * ld];
  return b;
}

__device__ __forceinline__ v16bf load_b_f32(const float* src, int ld) {
  const int lane = threadIdx.x & 31;
  const int n  = lane & 15;
  const int kb = (lane >> 4) * 16;
  const float* p = src + (size_t)kb * ld + n;
  v16bf b;
#pragma unroll
  for (int e = 0; e < 16; ++e) b[e] = (__bf16)p[(size_t)e * ld];
  return b;
}

// B fragment of M^T where M is row-major [N x K]: b[e] = M[col0+n][kb+e] (contiguous).
__device__ __forceinline__ v16bf load_bT_bf16(const __bf16* rowmajor, int ld, int col0) {
  const int lane = threadIdx.x & 31;
  const int n  = lane & 15;
  const int kb = (lane >> 4) * 16;
  const __bf16* p = rowmajor + (size_t)(col0 + n) * ld + kb;
  v16bf b;
#pragma unroll
  for (int e = 0; e < 16; ++e) b[e] = p[e];
  return b;
}

// ---------- async staging: 4KB (64x32 bf16) global chunk -> LDS ----------
__device__ __forceinline__ void stage_chunk(const __bf16* __restrict__ g,
                                            __bf16* l, int tid) {
#if HAVE_ASYNC_LDS
  gv2ip gp = (gv2ip)(uint64_t*)(const_cast<__bf16*>(g));
  lv2ip lp = (lv2ip)(uint64_t*)l;
  __builtin_amdgcn_global_load_async_to_lds_b64(gp + tid,       lp + tid,       0, 0);
  __builtin_amdgcn_global_load_async_to_lds_b64(gp + 256 + tid, lp + 256 + tid, 0, 0);
#else
  const uint64_t* gs = (const uint64_t*)g;
  uint64_t* ls = (uint64_t*)l;
  ls[tid]       = gs[tid];
  ls[256 + tid] = gs[256 + tid];
#endif
}

__device__ __forceinline__ void async_wait_all() {
#if HAVE_ASYNC_LDS
#if __has_builtin(__builtin_amdgcn_s_wait_asynccnt)
  __builtin_amdgcn_s_wait_asynccnt(0);
#else
  asm volatile("s_wait_asynccnt 0x0" ::: "memory");
#endif
#endif
}

// ---------- kernel 1: fused QKV projection (bf16 out, Q pre-scaled) ----------
// grid.x = (B*T)/64, grid.y = 3*H, block = 128 (4 waves, 16 rows each)
__global__ __launch_bounds__(128) void qkv_kernel(
    const float* __restrict__ x, const float* __restrict__ Wq,
    const float* __restrict__ Wk, const float* __restrict__ Wv,
    __bf16* __restrict__ Qw, __bf16* __restrict__ Kw, __bf16* __restrict__ Vw) {
  const int wave = threadIdx.x >> 5;
  const int g0   = blockIdx.x * 64 + wave * 16;   // row in [0, B*T)
  const int h    = blockIdx.y / 3;
  const int mat  = blockIdx.y % 3;                // 0=q, 1=k, 2=v
  const float* W = (mat == 0 ? Wq : (mat == 1 ? Wk : Wv)) + (size_t)h * C_ * HS_;
  __bf16* Ow     = (mat == 0 ? Qw : (mat == 1 ? Kw : Vw));

  v8f c0 = {}, c1 = {};
  for (int k0 = 0; k0 < C_; k0 += 32) {
    v16bf a  = load_a_f32(x + (size_t)g0 * C_ + k0, C_);
    v16bf b0 = load_b_f32(W + (size_t)k0 * HS_ + 0,  HS_);
    v16bf b1 = load_b_f32(W + (size_t)k0 * HS_ + 16, HS_);
    c0 = WMMA_BF16(a, b0, c0);
    c1 = WMMA_BF16(a, b1, c1);
  }
  const float scl = (mat == 0) ? 0.17677669529663687f : 1.0f;  // 1/sqrt(HS)

  const int b  = g0 / T_;
  const int t0 = g0 % T_;
  __bf16* out  = Ow + ((size_t)(b * H_ + h) * T_ + t0) * HS_;
  const int lane = threadIdx.x & 31;
  const int n  = lane & 15;
  const int rb = (lane >> 4) * 8;
#pragma unroll
  for (int r = 0; r < 8; ++r) {
    out[(size_t)(rb + r) * HS_ + n]      = (__bf16)(c0[r] * scl);
    out[(size_t)(rb + r) * HS_ + 16 + n] = (__bf16)(c1[r] * scl);
  }
}

// ---------- kernel 2: flash attention, async double-buffered K/V ----------
// grid.x = B*H, grid.y = T/128, block = 256 (8 waves, 16 query rows each)
__global__ __launch_bounds__(256) void attn_kernel(
    const __bf16* __restrict__ Qw, const __bf16* __restrict__ Kw,
    const __bf16* __restrict__ Vw, __bf16* __restrict__ Att) {
  __shared__ __align__(16) __bf16 Ks[2][SK_ * HS_];   // 2 x 4KB
  __shared__ __align__(16) __bf16 Vs[2][SK_ * HS_];   // 2 x 4KB
  __shared__ __align__(16) __bf16 Ps[8][16 * SK_];    // 16KB

  const int bh   = blockIdx.x;
  const int b    = bh / H_;
  const int h    = bh % H_;
  const int tq0  = blockIdx.y * 128;
  const int wave = threadIdx.x >> 5;
  const int lane = threadIdx.x & 31;
  const int tid  = threadIdx.x;

  const __bf16* Qbase = Qw + ((size_t)bh * T_ + tq0 + wave * 16) * HS_;
  const __bf16* Kbase = Kw + (size_t)bh * T_ * HS_;
  const __bf16* Vbase = Vw + (size_t)bh * T_ * HS_;

  const v16bf qa = load_a_bf16(Qbase, HS_);  // Q tile 16 x 32, resident all loop

  v8f o0 = {}, o1 = {};
  float mrow[8], lrow[8];
#pragma unroll
  for (int r = 0; r < 8; ++r) { mrow[r] = -3.0e38f; lrow[r] = 0.0f; }

  const int rb = (lane >> 4) * 8;
  const int n  = lane & 15;

  // prologue: stage chunk 0
  stage_chunk(Kbase, &Ks[0][0], tid);
  stage_chunk(Vbase, &Vs[0][0], tid);

  int cur = 0;
  for (int s0 = 0; s0 < T_; s0 += SK_) {
    async_wait_all();
    __syncthreads();                       // buffer `cur` ready for all waves

    if (s0 + SK_ < T_) {                   // overlap: stage next chunk
      stage_chunk(Kbase + (size_t)(s0 + SK_) * HS_, &Ks[cur ^ 1][0], tid);
      stage_chunk(Vbase + (size_t)(s0 + SK_) * HS_, &Vs[cur ^ 1][0], tid);
    }

    const __bf16* Kc = &Ks[cur][0];        // [64 keys][32 dims] row-major
    const __bf16* Vc = &Vs[cur][0];

    // S tiles: 16 q-rows x 64 keys = 4 WMMAs (K-dim = HS = 32)
    v8f sc[4];
#pragma unroll
    for (int j = 0; j < 4; ++j) {
      v8f z = {};
      sc[j] = WMMA_BF16(qa, load_bT_bf16(Kc, HS_, 16 * j), z);
    }

    // online softmax: one 4-step xor reduction per row per stat over 64 keys
    float pr[4][8], alpha[8];
#pragma unroll
    for (int r = 0; r < 8; ++r) {
      float mx = fmaxf(fmaxf(sc[0][r], sc[1][r]), fmaxf(sc[2][r], sc[3][r]));
#pragma unroll
      for (int off = 1; off < 16; off <<= 1) mx = fmaxf(mx, __shfl_xor(mx, off, 32));
      const float mnew = fmaxf(mrow[r], mx);
      alpha[r] = __expf(mrow[r] - mnew);
      mrow[r]  = mnew;
      float sum = 0.0f;
#pragma unroll
      for (int j = 0; j < 4; ++j) { pr[j][r] = __expf(sc[j][r] - mnew); sum += pr[j][r]; }
#pragma unroll
      for (int off = 1; off < 16; off <<= 1) sum += __shfl_xor(sum, off, 32);
      lrow[r] = lrow[r] * alpha[r] + sum;
    }

    // C-layout -> A-layout for P via per-wave LDS tile (16 x 64 bf16)
#pragma unroll
    for (int r = 0; r < 8; ++r) {
#pragma unroll
      for (int j = 0; j < 4; ++j)
        Ps[wave][(rb + r) * SK_ + 16 * j + n] = (__bf16)pr[j][r];
    }
    v16bf pa0 = load_a_bf16(&Ps[wave][0],      SK_);  // keys 0..31
    v16bf pa1 = load_a_bf16(&Ps[wave][0] + 32, SK_);  // keys 32..63

    v16bf vb00 = load_b_bf16(Vc,                 HS_);  // keys 0..31,  d 0..15
    v16bf vb01 = load_b_bf16(Vc + 16,            HS_);  // keys 0..31,  d 16..31
    v16bf vb10 = load_b_bf16(Vc + 32 * HS_,      HS_);  // keys 32..63, d 0..15
    v16bf vb11 = load_b_bf16(Vc + 32 * HS_ + 16, HS_);  // keys 32..63, d 16..31

#pragma unroll
    for (int r = 0; r < 8; ++r) { o0[r] *= alpha[r]; o1[r] *= alpha[r]; }
    o0 = WMMA_BF16(pa0, vb00, o0);
    o0 = WMMA_BF16(pa1, vb10, o0);
    o1 = WMMA_BF16(pa0, vb01, o1);
    o1 = WMMA_BF16(pa1, vb11, o1);

    cur ^= 1;
  }

  // normalize and store concat-head layout [B*T, H*HS] bf16
  __bf16* out = Att + ((size_t)b * T_ + tq0 + wave * 16) * (H_ * HS_) + h * HS_;
#pragma unroll
  for (int r = 0; r < 8; ++r) {
    const float inv = 1.0f / lrow[r];
    out[(size_t)(rb + r) * (H_ * HS_) + n]      = (__bf16)(o0[r] * inv);
    out[(size_t)(rb + r) * (H_ * HS_) + 16 + n] = (__bf16)(o1[r] * inv);
  }
}

// ---------- kernel 3: output projection + bias (f32 out) ----------
// grid.x = (B*T)/64, grid.y = C/32, block = 128
__global__ __launch_bounds__(128) void proj_kernel(
    const __bf16* __restrict__ Att, const float* __restrict__ Wproj,
    const float* __restrict__ bproj, float* __restrict__ out) {
  const int wave = threadIdx.x >> 5;
  const int g0   = blockIdx.x * 64 + wave * 16;
  const int col0 = blockIdx.y * 32;

  v8f c0 = {}, c1 = {};
  for (int k0 = 0; k0 < C_; k0 += 32) {
    v16bf a  = load_a_bf16(Att + (size_t)g0 * C_ + k0, C_);
    v16bf b0 = load_b_f32(Wproj + (size_t)k0 * C_ + col0,      C_);
    v16bf b1 = load_b_f32(Wproj + (size_t)k0 * C_ + col0 + 16, C_);
    c0 = WMMA_BF16(a, b0, c0);
    c1 = WMMA_BF16(a, b1, c1);
  }
  const int lane = threadIdx.x & 31;
  const int n  = lane & 15;
  const int rb = (lane >> 4) * 8;
  const float bias0 = bproj[col0 + n];
  const float bias1 = bproj[col0 + 16 + n];
#pragma unroll
  for (int r = 0; r < 8; ++r) {
    out[(size_t)(g0 + rb + r) * C_ + col0 + n]      = c0[r] + bias0;
    out[(size_t)(g0 + rb + r) * C_ + col0 + 16 + n] = c1[r] + bias1;
  }
}

extern "C" void kernel_launch(void* const* d_in, const int* in_sizes, int n_in,
                              void* d_out, int out_size, void* d_ws, size_t ws_size,
                              hipStream_t stream) {
  (void)in_sizes; (void)n_in; (void)out_size; (void)ws_size;
  const float* x  = (const float*)d_in[0];
  const float* Wq = (const float*)d_in[1];
  const float* Wk = (const float*)d_in[2];
  const float* Wv = (const float*)d_in[3];
  const float* Wp = (const float*)d_in[4];
  const float* bp = (const float*)d_in[5];
  float* out = (float*)d_out;

  const size_t qkvElems = (size_t)B_ * H_ * T_ * HS_;  // 3,145,728 each
  __bf16* Qw  = (__bf16*)d_ws;
  __bf16* Kw  = Qw + qkvElems;
  __bf16* Vw  = Kw + qkvElems;
  __bf16* Att = Vw + qkvElems;                          // [B*T, H*HS] bf16
  // total workspace: 4 * 3,145,728 * 2B = 25,165,824 bytes

  qkv_kernel<<<dim3((B_ * T_) / 64, 3 * H_), 128, 0, stream>>>(x, Wq, Wk, Wv, Qw, Kw, Vw);
  attn_kernel<<<dim3(B_ * H_, T_ / 128), 256, 0, stream>>>(Qw, Kw, Vw, Att);
  proj_kernel<<<dim3((B_ * T_) / 64, C_ / 32), 128, 0, stream>>>(Att, Wp, bp, out);
}
